// SS2D_1_3092376453151
// MI455X (gfx1250) — matrix operand order repaired and checked by reading, not compiled, not run on hardware
//
#include <hip/hip_runtime.h>

typedef __attribute__((ext_vector_type(16))) __bf16 v16bf;
typedef __attribute__((ext_vector_type(8)))  float  v8f;

static constexpr int cB  = 2;
static constexpr int cH  = 64;
static constexpr int cW  = 64;
static constexpr int cDM = 192;
static constexpr int cDI = 384;
static constexpr int cN  = 16;
static constexpr int cR  = 12;
static constexpr int cK  = 4;
static constexpr int cL  = 4096;     // H*W
static constexpr int cM  = cB * cL;  // 8192 rows through the big GEMMs
static constexpr int cCP = 48;       // x_dbl padded column count (44 -> 48)
static constexpr int cCH = 64;       // scan chunk length
static constexpr int cNC = cL / cCH; // 64 chunks

// ---------------------------------------------------------------------------
// helpers
// ---------------------------------------------------------------------------
__device__ __forceinline__ unsigned short f32_to_bf16(float f) {
  unsigned int u = __float_as_uint(f);
  u += 0x7FFFu + ((u >> 16) & 1u);   // round-to-nearest-even
  return (unsigned short)(u >> 16);
}

// A-fragment k-offset within a 32-wide K chunk (16x16x32 bf16 A layout)
__device__ __forceinline__ int a_koff(int half, int e) {
  return (e < 8) ? (8 * half + e) : (16 + 8 * half + (e - 8));
}

// cross-scan permutation: scan index t (direction k) -> spatial row-major idx
__device__ __forceinline__ int perm_idx(int k, int t) {
  int tt = (k >= 2) ? (cL - 1 - t) : t;
  if (k & 1) return ((tt & 63) << 6) | (tt >> 6);   // transpose (H=W=64)
  return tt;
}

__device__ __forceinline__ float silu_f(float v) {
  return v * (1.0f / (1.0f + __expf(-v)));
}

// ---------------------------------------------------------------------------
// pack B operand (weights): src is (N,K) row-major fp32, dst is WMMA-fragment
// ordered bf16: (((nt*kc + kci)*32 + lane)*16 + e), B k-off = a_koff(half, e)
// ---------------------------------------------------------------------------
__global__ void pack_b_kernel(const float* __restrict__ src,
                              unsigned short* __restrict__ dst,
                              int Nreal, int Npad, int Kt, int batches,
                              long srcBatchStride) {
  long idx = (long)blockIdx.x * blockDim.x + threadIdx.x;
  int kc = Kt >> 5;
  int ntiles = Npad >> 4;
  long perBatch = (long)ntiles * kc * 512;
  if (idx >= perBatch * batches) return;
  int  batch = (int)(idx / perBatch);
  long rl    = idx % perBatch;
  int  e     = (int)(rl & 15);
  int  lane  = (int)((rl >> 4) & 31);
  long tp    = rl >> 9;
  int  kci   = (int)(tp % kc);
  int  nt    = (int)(tp / kc);
  int  half  = lane >> 4;
  int  n     = nt * 16 + (lane & 15);
  int  k     = kci * 32 + a_koff(half, e);
  float v = (n < Nreal) ? src[(long)batch * srcBatchStride + (long)n * Kt + k] : 0.0f;
  dst[idx] = f32_to_bf16(v);
}

// pack A operand (activations): src (M,Kt) row-major fp32 -> fragment bf16
__global__ void pack_a_plain(const float* __restrict__ src,
                             unsigned short* __restrict__ dst, int M, int Kt) {
  long idx = (long)blockIdx.x * blockDim.x + threadIdx.x;
  long total = (long)M * Kt;
  if (idx >= total) return;
  int kc   = Kt >> 5;
  int e    = (int)(idx & 15);
  int lane = (int)((idx >> 4) & 31);
  long tp  = idx >> 9;
  int kci  = (int)(tp % kc);
  int mt   = (int)(tp / kc);
  int m    = mt * 16 + (lane & 15);
  int k    = kci * 32 + a_koff(lane >> 4, e);
  dst[idx] = f32_to_bf16(src[(long)m * Kt + k]);
}

// cross-scan + bf16 fragment pack: xs[b,k,t,d] = xc[b, perm_k(t), d]
__global__ void pack_xs_kernel(const float* __restrict__ xc,
                               unsigned short* __restrict__ dst) {
  long idx = (long)blockIdx.x * blockDim.x + threadIdx.x;
  const long perBatch = (long)cL * cDI;           // 1572864
  if (idx >= perBatch * (cB * cK)) return;
  int  batch = (int)(idx / perBatch);
  long rl    = idx % perBatch;
  int  e     = (int)(rl & 15);
  int  lane  = (int)((rl >> 4) & 31);
  long tp    = rl >> 9;
  const int kc = cDI >> 5;                        // 12
  int  kci   = (int)(tp % kc);
  int  mt    = (int)(tp / kc);
  int  t     = mt * 16 + (lane & 15);
  int  d     = kci * 32 + a_koff(lane >> 4, e);
  int  b     = batch >> 2;
  int  k     = batch & 3;
  int  p     = perm_idx(k, t);
  dst[idx] = f32_to_bf16(xc[((long)b * cL + p) * cDI + d]);
}

// ---------------------------------------------------------------------------
// WMMA GEMM on pre-packed bf16 fragments, f32 accumulate, software-pipelined:
// fragments for k-chunk kci+1 are loaded before the WMMAs of kci issue so
// VMEM overlaps the matrix pipe instead of s_wait_loadcnt 0 per WMMA.
// ---------------------------------------------------------------------------
template <int NTW>
__global__ void wmma_gemm_packed(const unsigned short* __restrict__ Apk,
                                 const unsigned short* __restrict__ Bpk,
                                 float* __restrict__ C,
                                 int M, int N, int Kt,
                                 int batches, long aStride, long bStride,
                                 int bMod, long cStride, int ldc) {
  int wave = (int)(((long)blockIdx.x * blockDim.x + threadIdx.x) >> 5);
  int lane = threadIdx.x & 31;
  int mtiles = M >> 4;
  int ngrp   = N / (16 * NTW);
  int wavesPerBatch = mtiles * ngrp;
  int batch = wave / wavesPerBatch;
  if (batch >= batches) return;
  int r  = wave % wavesPerBatch;
  int mt = r % mtiles;
  int ng = r / mtiles;
  int kc = Kt >> 5;

  const v16bf* Ab = (const v16bf*)(Apk + (long)batch * aStride);
  const v16bf* Bb = (const v16bf*)(Bpk + (long)(batch % bMod) * bStride);
  long aBase = ((long)mt * kc) * 32 + lane;

  v8f acc[NTW];
#pragma unroll
  for (int i = 0; i < NTW; ++i) acc[i] = v8f{0, 0, 0, 0, 0, 0, 0, 0};

  // prologue: fragments for kci = 0
  v16bf a0 = Ab[aBase];
  v16bf b0[NTW];
#pragma unroll
  for (int i = 0; i < NTW; ++i)
    b0[i] = Bb[((long)(ng * NTW + i) * kc) * 32 + lane];

  for (int kci = 0; kci < kc - 1; ++kci) {
    // issue next-iteration loads first
    v16bf a1 = Ab[aBase + (long)(kci + 1) * 32];
    v16bf b1[NTW];
#pragma unroll
    for (int i = 0; i < NTW; ++i)
      b1[i] = Bb[((long)(ng * NTW + i) * kc + (kci + 1)) * 32 + lane];
    // compute on current fragments while loads are in flight
#pragma unroll
    for (int i = 0; i < NTW; ++i)
      acc[i] = __builtin_amdgcn_wmma_f32_16x16x32_bf16(
          false, a0, false, b0[i], (short)0, acc[i], false, false);
    a0 = a1;
#pragma unroll
    for (int i = 0; i < NTW; ++i) b0[i] = b1[i];
  }
  // epilogue
#pragma unroll
  for (int i = 0; i < NTW; ++i)
    acc[i] = __builtin_amdgcn_wmma_f32_16x16x32_bf16(
        false, a0, false, b0[i], (short)0, acc[i], false, false);

  int half = lane >> 4, cidx = lane & 15;
  int m0 = mt * 16;
  float* Cb = C + (long)batch * cStride;
#pragma unroll
  for (int i = 0; i < NTW; ++i) {
    int n = (ng * NTW + i) * 16 + cidx;
#pragma unroll
    for (int v = 0; v < 8; ++v) {
      int m = m0 + v + 8 * half;
      Cb[(long)m * ldc + n] = acc[i][v];
    }
  }
}

// ---------------------------------------------------------------------------
// depthwise 3x3 conv + bias + SiLU:  xz cols [0,384) -> xc (B,L,DI)
// ---------------------------------------------------------------------------
__global__ void conv_silu_kernel(const float* __restrict__ xz,
                                 const float* __restrict__ cw,
                                 const float* __restrict__ cb,
                                 float* __restrict__ xc) {
  long idx = (long)blockIdx.x * blockDim.x + threadIdx.x;
  if (idx >= (long)cB * cL * cDI) return;
  int d  = (int)(idx % cDI);
  long bl = idx / cDI;
  int l  = (int)(bl & (cL - 1));
  int b  = (int)(bl >> 12);
  int h  = l >> 6, w = l & 63;
  float acc = cb[d];
#pragma unroll
  for (int ky = 0; ky < 3; ++ky) {
    int hh = h + ky - 1;
    if (hh < 0 || hh >= cH) continue;
#pragma unroll
    for (int kx = 0; kx < 3; ++kx) {
      int ww = w + kx - 1;
      if (ww < 0 || ww >= cW) continue;
      acc += cw[d * 9 + ky * 3 + kx] *
             xz[((long)b * cL + ((hh << 6) + ww)) * (2 * cDI) + d];
    }
  }
  xc[idx] = silu_f(acc);
}

// ---------------------------------------------------------------------------
// dt projection + softplus:  delta[bk,t,d] = softplus(x_dbl[bk,t,0:12]·Wdt + bias)
// ---------------------------------------------------------------------------
__global__ void dt_softplus_kernel(const float* __restrict__ xdbl,
                                   const float* __restrict__ dtw,
                                   const float* __restrict__ dtb,
                                   float* __restrict__ delta) {
  long idx = (long)blockIdx.x * blockDim.x + threadIdx.x;
  if (idx >= (long)cB * cK * cL * cDI) return;
  int d   = (int)(idx % cDI);
  long r2 = idx / cDI;
  int t   = (int)(r2 & (cL - 1));
  int bk  = (int)(r2 >> 12);
  int k   = bk & 3;
  const float* row  = xdbl + ((long)bk * cL + t) * cCP;
  const float* wrow = dtw + ((long)k * cDI + d) * cR;
  float acc = dtb[k * cDI + d];
#pragma unroll
  for (int r = 0; r < cR; ++r) acc += row[r] * wrow[r];
  delta[idx] = (acc > 20.0f) ? acc : log1pf(__expf(acc));
}

// ---------------------------------------------------------------------------
// Chunked parallel selective scan.  Recurrence h_t = a_t*h_{t-1} + b_t with
// a_t = exp(dt*A), b_t = dt*u*B.  Pass 1: per-chunk (prod a, reduced b) fully
// parallel over (bk,d,n,chunk).  Pass 2: 64-step serial combine per state
// lane -> per-chunk initial h.  Pass 3: parallel chunk replay with outputs.
// ---------------------------------------------------------------------------
__global__ void scan_pass1(const float* __restrict__ xc,
                           const float* __restrict__ xdbl,
                           const float* __restrict__ delta,
                           const float* __restrict__ A_logs,
                           float* __restrict__ aprod,
                           float* __restrict__ bstate) {
  int wid  = threadIdx.x >> 5;
  int lane = threadIdx.x & 31;
  int chunk = blockIdx.x & (cNC - 1);
  int rest  = blockIdx.x >> 6;
  int bk   = rest / 24;
  int dblk = rest % 24;
  int b = bk >> 2, k = bk & 3;
  int d = dblk * 16 + wid * 2 + (lane >> 4);
  int n = lane & 15;
  float An = -__expf(A_logs[(k * cDI + d) * cN + n]);

  const float* dbase = delta + (long)bk * cL * cDI + d;
  const float* xdb   = xdbl + (long)bk * cL * cCP;
  const float* xcb   = xc + (long)b * cL * cDI + d;

  float ar = 1.0f, br = 0.0f;
  int t0 = chunk * cCH;
  for (int tt = 0; tt < cCH; ++tt) {
    int t = t0 + tt;
    if ((tt & 7) == 0) {
      __builtin_prefetch(&dbase[(long)(t + 8) * cDI], 0, 0);
      __builtin_prefetch(&xdb[(long)(t + 8) * cCP], 0, 0);
    }
    int   p  = perm_idx(k, t);
    float dt = dbase[(long)t * cDI];
    float u  = xcb[(long)p * cDI];
    float Bv = xdb[(long)t * cCP + cR + n];
    float dA = __expf(dt * An);
    ar *= dA;
    br = br * dA + (dt * u) * Bv;
  }
  long sidx = ((((long)bk * cDI + d) * cN + n) * cNC) + chunk;
  aprod[sidx]  = ar;
  bstate[sidx] = br;
}

__global__ void scan_pass2(const float* __restrict__ aprod,
                           float* __restrict__ bstate /* -> h_init */) {
  long idx = (long)blockIdx.x * blockDim.x + threadIdx.x;  // (bk,d,n)
  if (idx >= (long)cB * cK * cDI * cN) return;
  const float* ap = aprod + idx * cNC;
  float*       bs = bstate + idx * cNC;
  float h = 0.0f;
  for (int c = 0; c < cNC; ++c) {
    float a = ap[c], bv = bs[c];
    bs[c] = h;                 // state entering chunk c
    h = h * a + bv;
  }
}

__global__ void scan_pass3(const float* __restrict__ xc,
                           const float* __restrict__ xdbl,
                           const float* __restrict__ delta,
                           const float* __restrict__ A_logs,
                           const float* __restrict__ Ds,
                           const float* __restrict__ hinit,
                           float* __restrict__ y) {
  int wid  = threadIdx.x >> 5;
  int lane = threadIdx.x & 31;
  int chunk = blockIdx.x & (cNC - 1);
  int rest  = blockIdx.x >> 6;
  int bk   = rest / 24;
  int dblk = rest % 24;
  int b = bk >> 2, k = bk & 3;
  int d = dblk * 16 + wid * 2 + (lane >> 4);
  int n = lane & 15;
  int ch = k * cDI + d;
  float An = -__expf(A_logs[ch * cN + n]);
  float Dv = Ds[ch];

  const float* dbase = delta + (long)bk * cL * cDI + d;
  const float* xdb   = xdbl + (long)bk * cL * cCP;
  const float* xcb   = xc + (long)b * cL * cDI + d;
  float*       yb    = y + (long)b * cL * cDI + d;

  float h = hinit[((((long)bk * cDI + d) * cN + n) * cNC) + chunk];
  int t0 = chunk * cCH;
  for (int tt = 0; tt < cCH; ++tt) {
    int t = t0 + tt;
    if ((tt & 7) == 0) {
      __builtin_prefetch(&dbase[(long)(t + 8) * cDI], 0, 0);
      __builtin_prefetch(&xdb[(long)(t + 8) * cCP], 0, 0);
    }
    int   p  = perm_idx(k, t);
    float dt = dbase[(long)t * cDI];
    float u  = xcb[(long)p * cDI];
    float Bv = xdb[(long)t * cCP + cR + n];
    float Cv = xdb[(long)t * cCP + cR + cN + n];
    h = h * __expf(dt * An) + (dt * u) * Bv;
    float acc = h * Cv;
    acc += __shfl_xor(acc, 1, 32);
    acc += __shfl_xor(acc, 2, 32);
    acc += __shfl_xor(acc, 4, 32);
    acc += __shfl_xor(acc, 8, 32);
    if (n == 0) unsafeAtomicAdd(yb + (long)p * cDI, acc + Dv * u);
  }
}

// ---------------------------------------------------------------------------
// LayerNorm over DI + SiLU gate, emitting packed bf16 A-fragments for out_proj
// ---------------------------------------------------------------------------
__global__ void norm_gate_pack_kernel(const float* __restrict__ y,
                                      const float* __restrict__ xz,
                                      const float* __restrict__ g,
                                      const float* __restrict__ bta,
                                      unsigned short* __restrict__ apk) {
  int bl = blockIdx.x;        // 0..8191 (row b*L + l)
  int tid = threadIdx.x;      // 0..383  (d)
  __shared__ float red[cDI];
  float v = y[(long)bl * cDI + tid];
  red[tid] = v;
  __syncthreads();
  if (tid < 128) red[tid] += red[tid + 256];
  __syncthreads();
  for (int s = 128; s > 0; s >>= 1) {
    if (tid < s) red[tid] += red[tid + s];
    __syncthreads();
  }
  float mu = red[0] * (1.0f / cDI);
  __syncthreads();
  float dv = v - mu;
  red[tid] = dv * dv;
  __syncthreads();
  if (tid < 128) red[tid] += red[tid + 256];
  __syncthreads();
  for (int s = 128; s > 0; s >>= 1) {
    if (tid < s) red[tid] += red[tid + s];
    __syncthreads();
  }
  float var = red[0] * (1.0f / cDI);
  float nrm = dv * rsqrtf(var + 1e-5f) * g[tid] + bta[tid];
  float z   = xz[(long)bl * (2 * cDI) + cDI + tid];
  float out = nrm * silu_f(z);
  // scatter into A-fragment layout (M=8192, Kt=384, kc=12)
  int mt = bl >> 4, rr = bl & 15, dd = tid & 31, kci = tid >> 5;
  int half = (dd >> 3) & 1;
  int e    = (dd & 7) + ((dd & 16) ? 8 : 0);
  int lane = half * 16 + rr;
  long idx = (((long)mt * 12 + kci) * 32 + lane) * 16 + e;
  apk[idx] = f32_to_bf16(out);
}

// ---------------------------------------------------------------------------
// host launcher
// ---------------------------------------------------------------------------
extern "C" void kernel_launch(void* const* d_in, const int* in_sizes, int n_in,
                              void* d_out, int out_size, void* d_ws, size_t ws_size,
                              hipStream_t stream) {
  (void)in_sizes; (void)n_in; (void)out_size;
  const float* x      = (const float*)d_in[0];
  const float* inW    = (const float*)d_in[1];
  const float* convW  = (const float*)d_in[2];
  const float* convB  = (const float*)d_in[3];
  const float* xprojW = (const float*)d_in[4];
  const float* dtW    = (const float*)d_in[5];
  const float* dtB    = (const float*)d_in[6];
  const float* Alogs  = (const float*)d_in[7];
  const float* Dsp    = (const float*)d_in[8];
  const float* normG  = (const float*)d_in[9];
  const float* normBt = (const float*)d_in[10];
  const float* outW   = (const float*)d_in[11];
  float* out = (float*)d_out;

  char* ws = (char*)d_ws;
  size_t off = 0;
  auto take = [&](size_t bytes) -> void* {
    void* p = ws + off;
    off = (off + bytes + 255) & ~(size_t)255;
    return p;
  };
  float* xz    = (float*)take((size_t)cM * (2 * cDI) * 4);            // 25.2 MB
  float* xc    = (float*)take((size_t)cM * cDI * 4);                  // 12.6 MB
  float* xdbl  = (float*)take((size_t)cB * cK * cL * cCP * 4);        //  6.3 MB
  float* delta = (float*)take((size_t)cB * cK * cL * cDI * 4);        // 50.3 MB
  float* ybuf  = (float*)take((size_t)cM * cDI * 4);                  // 12.6 MB
  float* aprod = (float*)take((size_t)cB * cK * cDI * cN * cNC * 4);  // 12.6 MB
  float* bstat = (float*)take((size_t)cB * cK * cDI * cN * cNC * 4);  // 12.6 MB
  unsigned short* Apk1 = (unsigned short*)take((size_t)cM * cDM * 2);
  unsigned short* Apk2 = (unsigned short*)take((size_t)cB * cK * cL * cDI * 2);
  unsigned short* Apk3 = (unsigned short*)take((size_t)cM * cDI * 2);
  unsigned short* Bpk1 = (unsigned short*)take((size_t)cDM * (2 * cDI) * 2);
  unsigned short* Bpk2 = (unsigned short*)take((size_t)cK * cDI * cCP * 2);
  unsigned short* Bpk3 = (unsigned short*)take((size_t)cDI * cDM * 2);
  if (off > ws_size) return;  // workspace too small: deterministic no-op

  const int TPB = 256;
  auto blks = [](long n, int t) { return (unsigned)((n + t - 1) / t); };

  // 1. pack weights to WMMA B-fragments (bf16)
  pack_b_kernel<<<blks((long)48 * 6 * 512, TPB), TPB, 0, stream>>>(
      inW, Bpk1, 2 * cDI, 2 * cDI, cDM, 1, 0);
  pack_b_kernel<<<blks((long)cK * 3 * 12 * 512, TPB), TPB, 0, stream>>>(
      xprojW, Bpk2, cR + 2 * cN, cCP, cDI, cK, (long)(cR + 2 * cN) * cDI);
  pack_b_kernel<<<blks((long)12 * 12 * 512, TPB), TPB, 0, stream>>>(
      outW, Bpk3, cDM, cDM, cDI, 1, 0);
  // 2. pack x to A-fragments
  pack_a_plain<<<blks((long)cM * cDM, TPB), TPB, 0, stream>>>(x, Apk1, cM, cDM);
  // 3. in_proj: xz = x @ in_proj_w^T   [8192x192]x[192x768]
  wmma_gemm_packed<4><<<blks((long)512 * 12 * 32, TPB), TPB, 0, stream>>>(
      Apk1, Bpk1, xz, cM, 2 * cDI, cDM, 1, 0, 0, 1, 0, 2 * cDI);
  // 4. depthwise conv + SiLU
  conv_silu_kernel<<<blks((long)cB * cL * cDI, TPB), TPB, 0, stream>>>(
      xz, convW, convB, xc);
  // 5. cross-scan + bf16 fragment pack
  pack_xs_kernel<<<blks((long)cB * cK * cL * cDI, TPB), TPB, 0, stream>>>(xc, Apk2);
  // 6. x_proj: x_dbl (batched over 8 (b,k)), N padded 44->48
  wmma_gemm_packed<3><<<blks((long)8 * 256 * 32, TPB), TPB, 0, stream>>>(
      Apk2, Bpk2, xdbl, cL, cCP, cDI,
      8, (long)cL * cDI, (long)cDI * cCP, cK, (long)cL * cCP, cCP);
  // 7. dt projection + softplus
  dt_softplus_kernel<<<blks((long)cB * cK * cL * cDI, TPB), TPB, 0, stream>>>(
      xdbl, dtW, dtB, delta);
  // 8. chunked parallel scan: pass1 (chunk transfer ops, machine-filling),
  //    pass2 (serial chunk combine), pass3 (chunk replay + atomic merge)
  hipMemsetAsync(ybuf, 0, (size_t)cM * cDI * 4, stream);
  scan_pass1<<<cB * cK * 24 * cNC, 256, 0, stream>>>(
      xc, xdbl, delta, Alogs, aprod, bstat);
  scan_pass2<<<blks((long)cB * cK * cDI * cN, TPB), TPB, 0, stream>>>(aprod, bstat);
  scan_pass3<<<cB * cK * 24 * cNC, 256, 0, stream>>>(
      xc, xdbl, delta, Alogs, Dsp, bstat, ybuf);
  // 9. LayerNorm + SiLU gate, emit packed bf16 A for out_proj
  norm_gate_pack_kernel<<<cM, cDI, 0, stream>>>(ybuf, xz, normG, normBt, Apk3);
  // 10. out_proj -> d_out   [8192x384]x[384x192]
  wmma_gemm_packed<4><<<blks((long)512 * 3 * 32, TPB), TPB, 0, stream>>>(
      Apk3, Bpk3, out, cM, cDM, cDI, 1, 0, 0, 1, 0, cDM);
}
